// MultiHeadAttention_41686952575626
// MI455X (gfx1250) — matrix operand order repaired and checked by reading, not compiled, not run on hardware
//
#include <hip/hip_runtime.h>

// ---------------------------------------------------------------------------
// MHA forward for MI455X (gfx1250, wave32, WMMA bf16).
//   DIM=1024, H=16, hd=64, B=2, S=2048.
//   Pipeline: cvt(x,W*)->bf16 | QKV GEMM (wmma) -> RoPE -> flash attention
//   (wmma + online softmax) -> output projection (wmma, f32 out).
// Workspace (bf16 elems): q|k|v|o (4M each) + xb (4M) + Wq|Wk|Wv|Wo (1M each)
//   = 24M elems = 48 MB.
// ---------------------------------------------------------------------------

typedef __bf16 bf16_t;
typedef __attribute__((ext_vector_type(16))) __bf16 v16bf;
typedef __attribute__((ext_vector_type(8)))  float  v8f;

#define DIM    1024
#define NH     16
#define HD     64
#define BATCH  2
#define SEQ    2048
#define MROWS  (BATCH * SEQ)                       // 4096
#define REGSZ  ((size_t)BATCH * NH * SEQ * HD)     // 4,194,304 bf16 elems

// ---- helpers --------------------------------------------------------------

struct U32B { uint4 lo; uint4 hi; };               // 32 bytes

__device__ __forceinline__ uint4 ld16B(const void* p) {
  return *(const uint4*)p;
}

// Two 16-byte chunks -> one v16bf fragment, no per-element moves.
__device__ __forceinline__ v16bf mk16(uint4 lo, uint4 hi) {
  U32B t{lo, hi};
  return __builtin_bit_cast(v16bf, t);
}

// Pack two f32 into two bf16 (hardware v_cvt, RTNE).
__device__ __forceinline__ unsigned pk2(float a, float b) {
  unsigned short ua = __builtin_bit_cast(unsigned short, (bf16_t)a);
  unsigned short ub = __builtin_bit_cast(unsigned short, (bf16_t)b);
  return (unsigned)ua | ((unsigned)ub << 16);
}

__device__ __forceinline__ v8f wmma_bf16(v16bf a, v16bf b, v8f c) {
  // D = A(16x32) * B(32x16) + C, f32 accumulate
  return __builtin_amdgcn_wmma_f32_16x16x32_bf16(
      false, a, false, b, (short)0, c, false, false);
}

// ---------------------------------------------------------------------------
// Kernel 0: f32 -> bf16 conversion, 8 elements/thread, b128 in / b128 out.
// ---------------------------------------------------------------------------
__global__ __launch_bounds__(256)
void cvt_kernel(const float* __restrict__ src, bf16_t* __restrict__ dst, int n8) {
  int i = blockIdx.x * 256 + threadIdx.x;
  if (i >= n8) return;
  const float4* s = (const float4*)src + (size_t)i * 2;
  float4 f0 = s[0], f1 = s[1];
  uint4 o;
  o.x = pk2(f0.x, f0.y);
  o.y = pk2(f0.z, f0.w);
  o.z = pk2(f1.x, f1.y);
  o.w = pk2(f1.z, f1.w);
  ((uint4*)dst)[i] = o;
}

// ---------------------------------------------------------------------------
// Kernel 1: q/k/v = x @ W^T (all bf16), stored bf16 in [B,H,S,hd] layout.
//   grid = (MROWS/64, DIM/256, 3), block = 256 (8 waves).
//   wave: 32x64 output tile (2 A-frags x 4 B-frags -> 8 wmma / K-step).
// ---------------------------------------------------------------------------
__global__ __launch_bounds__(256)
void qkv_kernel(const bf16_t* __restrict__ xb,
                const bf16_t* __restrict__ Wq,
                const bf16_t* __restrict__ Wk,
                const bf16_t* __restrict__ Wv,
                bf16_t* __restrict__ qo,
                bf16_t* __restrict__ ko,
                bf16_t* __restrict__ vo) {
  const bf16_t* W = (blockIdx.z == 0) ? Wq : (blockIdx.z == 1) ? Wk : Wv;
  bf16_t* out     = (blockIdx.z == 0) ? qo : (blockIdx.z == 1) ? ko : vo;

  const int wave = threadIdx.x >> 5;
  const int lane = threadIdx.x & 31;
  const int lm = lane & 15, lh = lane >> 4;

  const int m0 = blockIdx.x * 64 + (wave & 1) * 32;
  const int n0 = blockIdx.y * 256 + (wave >> 1) * 64;

  const bf16_t* xr0 = xb + (size_t)(m0 + lm) * DIM;        // A rows (M, M+16)
  const bf16_t* xr1 = xb + (size_t)(m0 + 16 + lm) * DIM;
  const bf16_t* wr0 = W + (size_t)(n0 + lm) * DIM;         // B rows (N tiles)
  const bf16_t* wr1 = W + (size_t)(n0 + 16 + lm) * DIM;
  const bf16_t* wr2 = W + (size_t)(n0 + 32 + lm) * DIM;
  const bf16_t* wr3 = W + (size_t)(n0 + 48 + lm) * DIM;

  v8f c00 = {}, c01 = {}, c02 = {}, c03 = {};
  v8f c10 = {}, c11 = {}, c12 = {}, c13 = {};

#pragma unroll 2
  for (int k = 0; k < DIM; k += 32) {
    __builtin_prefetch(wr0 + k + 256, 0, 1);               // global_prefetch_b8
    const int ka = k + lh * 16;                            // A chunk base
    const int kb = k + lh * 16;                            // B chunk base
    v16bf a0 = mk16(ld16B(xr0 + ka), ld16B(xr0 + ka + 8));
    v16bf a1 = mk16(ld16B(xr1 + ka), ld16B(xr1 + ka + 8));
    v16bf b0 = mk16(ld16B(wr0 + kb), ld16B(wr0 + kb + 8));
    v16bf b1 = mk16(ld16B(wr1 + kb), ld16B(wr1 + kb + 8));
    v16bf b2 = mk16(ld16B(wr2 + kb), ld16B(wr2 + kb + 8));
    v16bf b3 = mk16(ld16B(wr3 + kb), ld16B(wr3 + kb + 8));
    c00 = wmma_bf16(a0, b0, c00);  c01 = wmma_bf16(a0, b1, c01);
    c02 = wmma_bf16(a0, b2, c02);  c03 = wmma_bf16(a0, b3, c03);
    c10 = wmma_bf16(a1, b0, c10);  c11 = wmma_bf16(a1, b1, c11);
    c12 = wmma_bf16(a1, b2, c12);  c13 = wmma_bf16(a1, b3, c13);
  }

  // C layout: lane holds C[m = r + 8*lh][n = lm]; scatter into [B,H,S,hd].
  v8f acc[2][4] = {{c00, c01, c02, c03}, {c10, c11, c12, c13}};
#pragma unroll
  for (int i = 0; i < 2; ++i) {
#pragma unroll
    for (int j = 0; j < 4; ++j) {
      const int nn = n0 + j * 16 + lm;
      const int h = nn >> 6, d = nn & 63;
#pragma unroll
      for (int r = 0; r < 8; ++r) {
        const int mm = m0 + i * 16 + 8 * lh + r;
        const int b = mm >> 11, s = mm & (SEQ - 1);
        out[((size_t)(b * NH + h) * SEQ + s) * HD + d] = (bf16_t)acc[i][j][r];
      }
    }
  }
}

// ---------------------------------------------------------------------------
// Kernel 2: RoPE in-place on q and k regions ([BH,S,hd] bf16).
// ---------------------------------------------------------------------------
__global__ void rope_kernel(bf16_t* __restrict__ q, bf16_t* __restrict__ k) {
  const int NP = BATCH * NH * SEQ * (HD / 2);              // 2^21
  int gi = blockIdx.x * blockDim.x + threadIdx.x;
  if (gi >= 2 * NP) return;
  bf16_t* reg = (gi < NP) ? q : k;
  int i = (gi < NP) ? gi : gi - NP;
  const int bh  = i >> 16;                                 // SEQ*32 = 65536
  const int rem = i & 65535;
  const int s = rem >> 5, d = rem & 31;
  const size_t base = ((size_t)bh * SEQ + s) * HD + d;
  // inv_freq[d] = 10000^(-d/32) = exp(-d * ln(10000)/32)
  const float ang = (float)s * __expf(-0.2878231366242557f * (float)d);
  float sn, cs;
  __sincosf(ang, &sn, &cs);
  const float x1 = (float)reg[base];
  const float x2 = (float)reg[base + 32];
  reg[base]      = (bf16_t)(x1 * cs - x2 * sn);
  reg[base + 32] = (bf16_t)(x2 * cs + x1 * sn);
}

// ---------------------------------------------------------------------------
// Kernel 3: causal flash attention.
//   grid = (SEQ/128, B*H), block = 256. Wave w handles 16 q-rows.
//   Per 32-key block: S = Q K^T (4 wmma), online softmax, O += P V (4 wmma).
// ---------------------------------------------------------------------------
__global__ __launch_bounds__(256)
void flash_kernel(const bf16_t* __restrict__ q,
                  const bf16_t* __restrict__ k,
                  const bf16_t* __restrict__ v,
                  bf16_t* __restrict__ o) {
  __shared__ bf16_t vt[HD][32];        // V^T tile: [d][key]          (4 KB)
  __shared__ bf16_t pb[8][16][32];     // per-wave P scratch [m][key] (8 KB)

  const int wave = threadIdx.x >> 5;
  const int lane = threadIdx.x & 31;
  const int lm = lane & 15, lh = lane >> 4;
  const int bh = blockIdx.y;
  const int b = bh >> 4, h = bh & 15;
  const int qBlk = blockIdx.x * 128;
  const int qRow = qBlk + wave * 16;

  // Preload Q A-fragments (reused across all key blocks).
  const bf16_t* qbase = q + ((size_t)bh * SEQ + qRow + lm) * HD;
  const v16bf qa0 = mk16(ld16B(qbase + lh * 16),      ld16B(qbase + lh * 16 + 8));
  const v16bf qa1 = mk16(ld16B(qbase + 32 + lh * 16), ld16B(qbase + 32 + lh * 16 + 8));

  v8f o0 = {}, o1 = {}, o2 = {}, o3 = {};
  float mr[8], lr[8];
#pragma unroll
  for (int r = 0; r < 8; ++r) { mr[r] = -3.0e38f; lr[r] = 0.0f; }

  const float scale = 0.125f;          // 1/sqrt(64)
  const int kEnd = qBlk + 128;         // uniform over the workgroup

  for (int kb = 0; kb < kEnd; kb += 32) {
    __syncthreads();
    {  // cooperatively stage V^T (32 keys x 64 dims) into LDS
      const int key = threadIdx.x >> 3;
      const int dd  = (threadIdx.x & 7) * 8;
      uint4 raw = ld16B(v + ((size_t)bh * SEQ + kb + key) * HD + dd);
      bf16_t vv[8];
      *(uint4*)vv = raw;
#pragma unroll
      for (int i = 0; i < 8; ++i) vt[dd + i][key] = vv[i];
    }
    __syncthreads();

    // ---- scores: S(16x32) = Q(16x64) K^T(64x32) --------------------------
    v8f s0 = {}, s1 = {};
    const bf16_t* kb0 = k + ((size_t)bh * SEQ + kb + lm) * HD;   // key tile 0
    const bf16_t* kb1 = kb0 + (size_t)16 * HD;                   // key tile 1
    v16bf kf;
    kf = mk16(ld16B(kb0 + lh * 16),      ld16B(kb0 + lh * 16 + 8));
    s0 = wmma_bf16(qa0, kf, s0);
    kf = mk16(ld16B(kb0 + 32 + lh * 16), ld16B(kb0 + 32 + lh * 16 + 8));
    s0 = wmma_bf16(qa1, kf, s0);
    kf = mk16(ld16B(kb1 + lh * 16),      ld16B(kb1 + lh * 16 + 8));
    s1 = wmma_bf16(qa0, kf, s1);
    kf = mk16(ld16B(kb1 + 32 + lh * 16), ld16B(kb1 + 32 + lh * 16 + 8));
    s1 = wmma_bf16(qa1, kf, s1);

    // ---- online softmax (row stats per C-VGPR index r) -------------------
    const int key0 = kb + lm, key1 = kb + 16 + lm;
    float p0[8], p1[8];
#pragma unroll
    for (int r = 0; r < 8; ++r) {
      const int qi = qRow + r + 8 * lh;
      float t0 = (key0 <= qi) ? s0[r] * scale : -3.0e38f;
      float t1 = (key1 <= qi) ? s1[r] * scale : -3.0e38f;
      float mx = fmaxf(t0, t1);
      mx = fmaxf(mx, __shfl_xor(mx, 1, 32));
      mx = fmaxf(mx, __shfl_xor(mx, 2, 32));
      mx = fmaxf(mx, __shfl_xor(mx, 4, 32));
      mx = fmaxf(mx, __shfl_xor(mx, 8, 32));
      const float mn = fmaxf(mr[r], mx);
      const float a0 = __expf(t0 - mn);
      const float a1 = __expf(t1 - mn);
      float rs = a0 + a1;
      rs += __shfl_xor(rs, 1, 32);
      rs += __shfl_xor(rs, 2, 32);
      rs += __shfl_xor(rs, 4, 32);
      rs += __shfl_xor(rs, 8, 32);
      const float al = __expf(mr[r] - mn);
      lr[r] = lr[r] * al + rs;
      mr[r] = mn;
      o0[r] *= al; o1[r] *= al; o2[r] *= al; o3[r] *= al;
      p0[r] = a0;  p1[r] = a1;
    }

    // ---- relayout P (C layout -> A fragment) via per-wave LDS ------------
#pragma unroll
    for (int r = 0; r < 8; ++r) {
      pb[wave][r + 8 * lh][lm]      = (bf16_t)p0[r];
      pb[wave][r + 8 * lh][16 + lm] = (bf16_t)p1[r];
    }
    asm volatile("" ::: "memory");      // keep ds stores before ds loads
    const v16bf pa = mk16(ld16B(&pb[wave][lm][lh * 16]),
                          ld16B(&pb[wave][lm][lh * 16 + 8]));
    __syncthreads();

    // ---- O(16x64) += P(16x32) V(32x64) -----------------------------------
    v16bf vf;
    vf = mk16(ld16B(&vt[lm][lh * 16]),      ld16B(&vt[lm][lh * 16 + 8]));
    o0 = wmma_bf16(pa, vf, o0);
    vf = mk16(ld16B(&vt[16 + lm][lh * 16]), ld16B(&vt[16 + lm][lh * 16 + 8]));
    o1 = wmma_bf16(pa, vf, o1);
    vf = mk16(ld16B(&vt[32 + lm][lh * 16]), ld16B(&vt[32 + lm][lh * 16 + 8]));
    o2 = wmma_bf16(pa, vf, o2);
    vf = mk16(ld16B(&vt[48 + lm][lh * 16]), ld16B(&vt[48 + lm][lh * 16 + 8]));
    o3 = wmma_bf16(pa, vf, o3);
  }

  // ---- epilogue: O /= l, store bf16 in [B,S,DIM] layout for out-proj -----
#pragma unroll
  for (int r = 0; r < 8; ++r) {
    const float inv = 1.0f / lr[r];
    const int s = qRow + r + 8 * lh;
    const size_t row = ((size_t)b * SEQ + s) * DIM + h * HD;
    o[row + lm]      = (bf16_t)(o0[r] * inv);
    o[row + 16 + lm] = (bf16_t)(o1[r] * inv);
    o[row + 32 + lm] = (bf16_t)(o2[r] * inv);
    o[row + 48 + lm] = (bf16_t)(o3[r] * inv);
  }
}

// ---------------------------------------------------------------------------
// Kernel 4: out = O @ Wo^T  (bf16 in, f32 out).
//   grid = (MROWS/64, DIM/256), block = 256; 32x64 tile per wave.
// ---------------------------------------------------------------------------
__global__ __launch_bounds__(256)
void oproj_kernel(const bf16_t* __restrict__ A,
                  const bf16_t* __restrict__ Wo,
                  float* __restrict__ out) {
  const int wave = threadIdx.x >> 5;
  const int lane = threadIdx.x & 31;
  const int lm = lane & 15, lh = lane >> 4;
  const int m0 = blockIdx.x * 64 + (wave & 1) * 32;
  const int n0 = blockIdx.y * 256 + (wave >> 1) * 64;

  const bf16_t* ar0 = A + (size_t)(m0 + lm) * DIM;
  const bf16_t* ar1 = A + (size_t)(m0 + 16 + lm) * DIM;
  const bf16_t* wr0 = Wo + (size_t)(n0 + lm) * DIM;
  const bf16_t* wr1 = Wo + (size_t)(n0 + 16 + lm) * DIM;
  const bf16_t* wr2 = Wo + (size_t)(n0 + 32 + lm) * DIM;
  const bf16_t* wr3 = Wo + (size_t)(n0 + 48 + lm) * DIM;

  v8f c00 = {}, c01 = {}, c02 = {}, c03 = {};
  v8f c10 = {}, c11 = {}, c12 = {}, c13 = {};

#pragma unroll 2
  for (int k = 0; k < DIM; k += 32) {
    __builtin_prefetch(wr0 + k + 256, 0, 1);
    const int ka = k + lh * 16;
    const int kb = k + lh * 16;
    v16bf a0 = mk16(ld16B(ar0 + ka), ld16B(ar0 + ka + 8));
    v16bf a1 = mk16(ld16B(ar1 + ka), ld16B(ar1 + ka + 8));
    v16bf b0 = mk16(ld16B(wr0 + kb), ld16B(wr0 + kb + 8));
    v16bf b1 = mk16(ld16B(wr1 + kb), ld16B(wr1 + kb + 8));
    v16bf b2 = mk16(ld16B(wr2 + kb), ld16B(wr2 + kb + 8));
    v16bf b3 = mk16(ld16B(wr3 + kb), ld16B(wr3 + kb + 8));
    c00 = wmma_bf16(a0, b0, c00);  c01 = wmma_bf16(a0, b1, c01);
    c02 = wmma_bf16(a0, b2, c02);  c03 = wmma_bf16(a0, b3, c03);
    c10 = wmma_bf16(a1, b0, c10);  c11 = wmma_bf16(a1, b1, c11);
    c12 = wmma_bf16(a1, b2, c12);  c13 = wmma_bf16(a1, b3, c13);
  }

  v8f acc[2][4] = {{c00, c01, c02, c03}, {c10, c11, c12, c13}};
#pragma unroll
  for (int i = 0; i < 2; ++i) {
#pragma unroll
    for (int j = 0; j < 4; ++j) {
      const int nn = n0 + j * 16 + lm;
#pragma unroll
      for (int r = 0; r < 8; ++r) {
        const size_t mm = (size_t)(m0 + i * 16 + 8 * lh + r);
        out[mm * DIM + nn] = acc[i][j][r];
      }
    }
  }
}

// ---------------------------------------------------------------------------
// Host launcher
// ---------------------------------------------------------------------------
extern "C" void kernel_launch(void* const* d_in, const int* in_sizes, int n_in,
                              void* d_out, int out_size, void* d_ws, size_t ws_size,
                              hipStream_t stream) {
  (void)in_sizes; (void)n_in; (void)out_size; (void)ws_size;
  const float* x  = (const float*)d_in[0];
  const float* Wq = (const float*)d_in[1];
  const float* Wk = (const float*)d_in[2];
  const float* Wv = (const float*)d_in[3];
  const float* Wo = (const float*)d_in[4];
  // d_in[5] = causal mask (implicit in flash kernel)

  bf16_t* ws = (bf16_t*)d_ws;
  bf16_t* qr  = ws;                                 // 4M elems
  bf16_t* kr  = ws + REGSZ;                         // 4M
  bf16_t* vr  = ws + 2 * REGSZ;                     // 4M
  bf16_t* orr = ws + 3 * REGSZ;                     // 4M
  bf16_t* xb  = ws + 4 * REGSZ;                     // 4M  (x in bf16)
  bf16_t* wqb = ws + 5 * REGSZ;                     // 1M each below
  bf16_t* wkb = wqb + (size_t)DIM * DIM;
  bf16_t* wvb = wkb + (size_t)DIM * DIM;
  bf16_t* wob = wvb + (size_t)DIM * DIM;

  const int NX8 = MROWS * DIM / 8;                  // 524,288
  const int NW8 = DIM * DIM / 8;                    // 131,072
  cvt_kernel<<<dim3((NX8 + 255) / 256), 256, 0, stream>>>(x,  xb,  NX8);
  cvt_kernel<<<dim3((NW8 + 255) / 256), 256, 0, stream>>>(Wq, wqb, NW8);
  cvt_kernel<<<dim3((NW8 + 255) / 256), 256, 0, stream>>>(Wk, wkb, NW8);
  cvt_kernel<<<dim3((NW8 + 255) / 256), 256, 0, stream>>>(Wv, wvb, NW8);
  cvt_kernel<<<dim3((NW8 + 255) / 256), 256, 0, stream>>>(Wo, wob, NW8);

  qkv_kernel<<<dim3(MROWS / 64, DIM / 256, 3), 256, 0, stream>>>(
      xb, wqb, wkb, wvb, qr, kr, vr);

  const int NP = BATCH * NH * SEQ * (HD / 2);
  rope_kernel<<<dim3((2 * NP + 255) / 256), 256, 0, stream>>>(qr, kr);

  flash_kernel<<<dim3(SEQ / 128, BATCH * NH), 256, 0, stream>>>(qr, kr, vr, orr);

  oproj_kernel<<<dim3(MROWS / 64, DIM / 256), 256, 0, stream>>>(
      orr, wob, (float*)d_out);
}
